// GuidedDiffusionBlock_82446192214392
// MI455X (gfx1250) — matrix operand order repaired and checked by reading, not compiled, not run on hardware
//
#include <hip/hip_runtime.h>

// ---------------- problem constants ----------------
#define Bsz 64
#define Nn  128
#define Dd  512
#define Cc  1024
#define Ee  131072
#define Rr  16
#define NBb 8
#define Ll  3
#define Hh  8
#define HDd 64
#define NN  (Bsz*Nn)        // 8192 nodes
#define SEGS (NN*Rr)        // 131072 segments

typedef unsigned short ushort_t;
typedef __attribute__((ext_vector_type(16))) __bf16 vbf16;
typedef __attribute__((ext_vector_type(8)))  __bf16 vbf8;
typedef __attribute__((ext_vector_type(8)))  float  vf8;
typedef __attribute__((ext_vector_type(4)))  float  vf4;
typedef __attribute__((ext_vector_type(4)))  unsigned int vu4;
typedef __attribute__((ext_vector_type(2)))  unsigned int vu2;

__device__ __forceinline__ unsigned short f2bf(float f) {
  unsigned int u = __builtin_bit_cast(unsigned int, f);
  unsigned int r = u + 0x7FFFu + ((u >> 16) & 1u);
  return (unsigned short)(r >> 16);
}
__device__ __forceinline__ float bf2f(unsigned int h) {
  return __builtin_bit_cast(float, h << 16);
}

__device__ __forceinline__ vf8 wmma_bf16(vbf16 a, vbf16 b, vf8 c) {
  return __builtin_amdgcn_wmma_f32_16x16x32_bf16(false, a, false, b, (short)0, c, false, false);
}

// CDNA5 async global->LDS copy (16B per lane), tracked by ASYNCcnt.
__device__ __forceinline__ void async_copy_b128(unsigned int lds_byte_addr, const void* gaddr) {
  asm volatile("global_load_async_to_lds_b128 %0, %1, off"
               :: "v"(lds_byte_addr), "v"(gaddr)
               : "memory");
}
__device__ __forceinline__ void wait_async0() {
  asm volatile("s_wait_asynccnt 0x0" ::: "memory");
}

// A fragment (16x32, bf16) from LDS, per CDNA5 ISA layout:
// lanes 0-15: row r0+lane, K halves {kb..kb+7, kb+16..kb+23} with kb=k0
// lanes 16-31: same rows, kb=k0+8
__device__ __forceinline__ vbf16 lds_a_frag(const ushort_t* base, int stride, int r0, int k0) {
  int lane = threadIdx.x & 31;
  int row = r0 + (lane & 15);
  int kb  = k0 + ((lane & 16) ? 8 : 0);
  const ushort_t* p = base + row * stride + kb;
  vu4 lo = *(const vu4*)p;
  vu4 hi = *(const vu4*)(p + 16);
  vbf8 a = __builtin_bit_cast(vbf8, lo);
  vbf8 b = __builtin_bit_cast(vbf8, hi);
  return __builtin_shufflevector(a, b, 0,1,2,3,4,5,6,7,8,9,10,11,12,13,14,15);
}
// B fragment (32x16, bf16): lane holds one column n, K halves {kb..kb+7, kb+16..kb+23}, kb=k0+8*(lane/16).
// LDS layout is column-major: Bl[n][k], stride in halves.
__device__ __forceinline__ vbf16 lds_b_frag(const ushort_t* base, int stride, int n0, int k0) {
  int lane = threadIdx.x & 31;
  int col = n0 + (lane & 15);
  int kb  = k0 + ((lane & 16) ? 8 : 0);
  const ushort_t* p = base + col * stride + kb;
  vu4 lo = *(const vu4*)p;
  vu4 hi = *(const vu4*)(p + 16);
  vbf8 a = __builtin_bit_cast(vbf8, lo);
  vbf8 b = __builtin_bit_cast(vbf8, hi);
  return __builtin_shufflevector(a, b, 0,1,2,3,4,5,6,7,8,9,10,11,12,13,14,15);
}

// ---------------- preprocessing: x + time-emb + maxpool(obj_cond) ----------------
__launch_bounds__(256)
__global__ void k_pre(const float* __restrict__ x, const int* __restrict__ t,
                      const float* __restrict__ obj,
                      float* __restrict__ h, ushort_t* __restrict__ hbf) {
  int row = blockIdx.x;                 // node 0..8191
  int b = row >> 7;
  float tv = (float)t[b];
  #pragma unroll
  for (int u = 0; u < 2; u++) {
    int dd = threadIdx.x + u * 256;
    float emb;
    if (dd < 256) {
      float f = __expf(-9.210340371976184f * (float)dd * (1.0f / 256.0f));
      emb = __sinf(tv * f);
    } else {
      float f = __expf(-9.210340371976184f * (float)(dd - 256) * (1.0f / 256.0f));
      emb = __cosf(tv * f);
    }
    float o0 = obj[(size_t)row * Cc + dd * 2];
    float o1 = obj[(size_t)row * Cc + dd * 2 + 1];
    float v = x[(size_t)row * Dd + dd] + emb + fmaxf(o0, o1);
    h[(size_t)row * Dd + dd] = v;
    hbf[(size_t)row * Dd + dd] = f2bf(v);
  }
}

__launch_bounds__(256)
__global__ void k_f2bf(const float* __restrict__ in, ushort_t* __restrict__ out, int n) {
  int i = blockIdx.x * 256 + threadIdx.x;
  if (i < n) out[i] = f2bf(in[i]);
}

__launch_bounds__(256)
__global__ void k_zero_i32(int* __restrict__ p, int n) {
  int i = blockIdx.x * 256 + threadIdx.x;
  if (i < n) p[i] = 0;
}

__launch_bounds__(256)
__global__ void k_zero_f32(float* __restrict__ p, int n) {
  int i = blockIdx.x * 256 + threadIdx.x;
  if (i < n) p[i] = 0.f;
}

// ---------------- CSR build over (dst, relation) segments ----------------
__launch_bounds__(256)
__global__ void k_count(const int* __restrict__ ec, const int* __restrict__ rel,
                        int* __restrict__ cnt) {
  int e = blockIdx.x * 256 + threadIdx.x;
  if (e >= Ee) return;
  int seg = ec[Ee + e] * Rr + rel[e];
  atomicAdd(&cnt[seg], 1);
}

__launch_bounds__(1024)
__global__ void k_scan(const int* __restrict__ cnt, int* __restrict__ offs, int n) {
  __shared__ int sh[1024];
  __shared__ int carry;
  int tid = threadIdx.x;
  if (tid == 0) carry = 0;
  __syncthreads();
  for (int base = 0; base < n; base += 1024) {
    int v = cnt[base + tid];
    sh[tid] = v;
    __syncthreads();
    for (int off = 1; off < 1024; off <<= 1) {
      int tv2 = (tid >= off) ? sh[tid - off] : 0;
      __syncthreads();
      sh[tid] += tv2;
      __syncthreads();
    }
    int incl = sh[tid];
    offs[base + tid] = carry + incl - v;   // exclusive
    __syncthreads();
    if (tid == 0) carry += sh[1023];
    __syncthreads();
  }
  if (tid == 0) offs[n] = carry;
}

__launch_bounds__(256)
__global__ void k_fill(const int* __restrict__ ec, const int* __restrict__ rel,
                       const int* __restrict__ offs, int* __restrict__ fill,
                       int* __restrict__ elist) {
  int e = blockIdx.x * 256 + threadIdx.x;
  if (e >= Ee) return;
  int seg = ec[Ee + e] * Rr + rel[e];
  int pos = atomicAdd(&fill[seg], 1);
  elist[offs[seg] + pos] = ec[e];          // src node
}

// sort each (short) segment list -> deterministic float summation order
__launch_bounds__(256)
__global__ void k_sortseg(const int* __restrict__ offs, int* __restrict__ elist) {
  int seg = blockIdx.x * 256 + threadIdx.x;
  if (seg >= SEGS) return;
  int beg = offs[seg], end = offs[seg + 1];
  for (int i = beg + 1; i < end; i++) {
    int v = elist[i];
    int j = i - 1;
    while (j >= beg && elist[j] > v) { elist[j + 1] = elist[j]; j--; }
    elist[j + 1] = v;
  }
}

// ---------------- per-layer: S[node][r*512+d] = mean of h[src] over segment ----------------
__launch_bounds__(256)
__global__ void k_aggregate(const float* __restrict__ h, const int* __restrict__ offs,
                            const int* __restrict__ elist, ushort_t* __restrict__ S) {
  int seg = blockIdx.x * 2 + (threadIdx.x >> 7);
  int tid = threadIdx.x & 127;
  int beg = offs[seg], end = offs[seg + 1];
  int col = tid * 4;
  float s0 = 0.f, s1 = 0.f, s2 = 0.f, s3 = 0.f;
  for (int i = beg; i < end; i++) {
    int sn = elist[i];
    vf4 v = *(const vf4*)(h + (size_t)sn * Dd + col);
    s0 += v[0]; s1 += v[1]; s2 += v[2]; s3 += v[3];
  }
  int c = end - beg;
  float ic = 1.0f / (float)(c > 0 ? c : 1);
  int node = seg >> 4, r = seg & 15;
  ushort_t* o = S + (size_t)node * (Rr * Dd) + r * Dd + col;
  vu2 w;
  w[0] = (unsigned int)f2bf(s0 * ic) | ((unsigned int)f2bf(s1 * ic) << 16);
  w[1] = (unsigned int)f2bf(s2 * ic) | ((unsigned int)f2bf(s3 * ic) << 16);
  *(vu2*)o = w;
}

// ---------------- Wcat[r*512+i][o] = sum_b comb[r,b]*basis[b,i,o] (bf16) ----------------
__launch_bounds__(256)
__global__ void k_wcat(const float* __restrict__ comb, const float* __restrict__ basis,
                       ushort_t* __restrict__ Wcat) {
  int idx = blockIdx.x * 256 + threadIdx.x;   // over 8192*512
  int o = idx & 511;
  int row = idx >> 9;
  int r = row >> 9, i = row & 511;
  float s = 0.f;
  #pragma unroll
  for (int bb = 0; bb < NBb; bb++)
    s += comb[r * NBb + bb] * basis[(size_t)bb * Dd * Dd + (size_t)i * Dd + o];
  Wcat[idx] = f2bf(s);
}

// ---------------- generic bf16 WMMA GEMM: C(f32)[M,N] (+)= A[M,K] @ B[K,N] ----------------
// 128x128 block tile, 8 wave32 waves each 32x64, K-step 32.
// Double-buffered LDS; A tile copied with CDNA5 async global->LDS (ASYNCcnt),
// B tile register-staged (transposed on the way into LDS).
#define A_LDH 40
#define B_LDH 40

template<int ACC, int ACT, int WBF>
__global__ __launch_bounds__(256)
void k_gemm(const ushort_t* __restrict__ A, const ushort_t* __restrict__ B,
            float* __restrict__ C, const float* __restrict__ bias,
            ushort_t* __restrict__ Cbf, int M, int N, int K) {
  __shared__ ushort_t Al[2][128 * A_LDH];
  __shared__ ushort_t Bl[2][128 * B_LDH];
  int m0 = blockIdx.y * 128;
  int n0 = blockIdx.x * 128;
  int t = threadIdx.x;
  int w = t >> 5;
  int wm = (w >> 1) * 32;
  int wn = (w & 1) * 64;

  // per-thread tile-copy coordinates (2 units of 16B each per tile)
  int a_row0 = t >> 2;                 int a_kp0 = (t & 3) * 8;
  int a_row1 = (t + 256) >> 2;         int a_kp1 = ((t + 256) & 3) * 8;
  int b_kr0 = t >> 4;                  int b_np0 = (t & 15) * 8;
  int b_kr1 = (t + 256) >> 4;          int b_np1 = ((t + 256) & 15) * 8;

  vf8 acc8[2][4];
  #pragma unroll
  for (int mi = 0; mi < 2; mi++)
    #pragma unroll
    for (int ni = 0; ni < 4; ni++)
      acc8[mi][ni] = (vf8){0.f,0.f,0.f,0.f,0.f,0.f,0.f,0.f};

  // ---- prologue: stage 0 ----
  async_copy_b128((unsigned int)(unsigned long long)(const void*)&Al[0][a_row0 * A_LDH + a_kp0],
                  (const void*)(A + (size_t)(m0 + a_row0) * K + a_kp0));
  async_copy_b128((unsigned int)(unsigned long long)(const void*)&Al[0][a_row1 * A_LDH + a_kp1],
                  (const void*)(A + (size_t)(m0 + a_row1) * K + a_kp1));
  {
    vu4 r0 = *(const vu4*)(B + (size_t)b_kr0 * N + n0 + b_np0);
    vu4 r1 = *(const vu4*)(B + (size_t)b_kr1 * N + n0 + b_np1);
    #pragma unroll
    for (int i = 0; i < 4; i++) {
      Bl[0][(b_np0 + 2*i)     * B_LDH + b_kr0] = (ushort_t)(r0[i] & 0xffffu);
      Bl[0][(b_np0 + 2*i + 1) * B_LDH + b_kr0] = (ushort_t)(r0[i] >> 16);
      Bl[0][(b_np1 + 2*i)     * B_LDH + b_kr1] = (ushort_t)(r1[i] & 0xffffu);
      Bl[0][(b_np1 + 2*i + 1) * B_LDH + b_kr1] = (ushort_t)(r1[i] >> 16);
    }
  }

  int cur = 0;
  int nsteps = K >> 5;
  for (int s = 0; s < nsteps; s++) {
    int k0 = s << 5;
    wait_async0();          // this wave's async writes into Al[cur] done
    __syncthreads();        // all waves' LDS writes for stage visible
    bool pf = (s + 1 < nsteps);
    vu4 rb0, rb1;
    if (pf) {
      int kn = k0 + 32;
      async_copy_b128((unsigned int)(unsigned long long)(const void*)&Al[cur ^ 1][a_row0 * A_LDH + a_kp0],
                      (const void*)(A + (size_t)(m0 + a_row0) * K + kn + a_kp0));
      async_copy_b128((unsigned int)(unsigned long long)(const void*)&Al[cur ^ 1][a_row1 * A_LDH + a_kp1],
                      (const void*)(A + (size_t)(m0 + a_row1) * K + kn + a_kp1));
      rb0 = *(const vu4*)(B + (size_t)(kn + b_kr0) * N + n0 + b_np0);
      rb1 = *(const vu4*)(B + (size_t)(kn + b_kr1) * N + n0 + b_np1);
    }
    // compute on current stage
    vbf16 af0 = lds_a_frag(Al[cur], A_LDH, wm,      0);
    vbf16 af1 = lds_a_frag(Al[cur], A_LDH, wm + 16, 0);
    #pragma unroll
    for (int ni = 0; ni < 4; ni++) {
      vbf16 bf = lds_b_frag(Bl[cur], B_LDH, wn + ni * 16, 0);
      acc8[0][ni] = wmma_bf16(af0, bf, acc8[0][ni]);
      acc8[1][ni] = wmma_bf16(af1, bf, acc8[1][ni]);
    }
    if (pf) {
      #pragma unroll
      for (int i = 0; i < 4; i++) {
        Bl[cur ^ 1][(b_np0 + 2*i)     * B_LDH + b_kr0] = (ushort_t)(rb0[i] & 0xffffu);
        Bl[cur ^ 1][(b_np0 + 2*i + 1) * B_LDH + b_kr0] = (ushort_t)(rb0[i] >> 16);
        Bl[cur ^ 1][(b_np1 + 2*i)     * B_LDH + b_kr1] = (ushort_t)(rb1[i] & 0xffffu);
        Bl[cur ^ 1][(b_np1 + 2*i + 1) * B_LDH + b_kr1] = (ushort_t)(rb1[i] >> 16);
      }
    }
    cur ^= 1;
  }

  // ---- epilogue (straight-line; flags are compile-time) ----
  int lane  = t & 31;
  int lhalf = (lane & 16) ? 8 : 0;
  int lcol  = lane & 15;
  #pragma unroll
  for (int mi = 0; mi < 2; mi++)
    #pragma unroll
    for (int ni = 0; ni < 4; ni++) {
      int col = n0 + wn + ni * 16 + lcol;
      float bv = bias[col];
      #pragma unroll
      for (int j = 0; j < 8; j++) {
        int row = m0 + wm + mi * 16 + j + lhalf;
        size_t o = (size_t)row * N + col;
        float v = acc8[mi][ni][j];
        if (ACC) v += C[o];
        v += bv;
        if (ACT) v = v / (1.0f + __expf(-v));   // SiLU
        C[o] = v;
        if (WBF) Cbf[o] = f2bf(v);
      }
    }
}

// ---------------- fused attention per (b,h): softmax(QK^T/8) V ----------------
__launch_bounds__(256)
__global__ void k_attn(const ushort_t* __restrict__ Q, const ushort_t* __restrict__ K,
                       const ushort_t* __restrict__ V, ushort_t* __restrict__ O) {
  // smem: [Ql 128x72 | Kl 128x72] unioned with [Pl 128x136]; Vt 64x136 after
  __shared__ ushort_t smem[18432 + 8704];
  ushort_t* Ql = smem;
  ushort_t* Kl = smem + 9216;
  ushort_t* Vt = smem + 18432;
  ushort_t* Pl = smem;            // reused after scores are consumed
  int bh = blockIdx.x;
  int b = bh >> 3, h = bh & 7;
  const size_t base = (size_t)b * Nn * (Hh * HDd) + (size_t)h * HDd;
  int t = threadIdx.x;
  // load Q (scaled by 1/8), K, V(transposed)
  #pragma unroll
  for (int u = 0; u < 4; u++) {
    int idx = t + u * 256;        // 1024 units over 128x64 halves
    int row = idx >> 3;
    int cp  = (idx & 7) * 8;
    size_t g = base + (size_t)row * (Hh * HDd) + cp;
    vu4 q = *(const vu4*)(Q + g);
    #pragma unroll
    for (int i = 0; i < 4; i++) {
      float lo = bf2f(q[i] & 0xffffu) * 0.125f;
      float hi = bf2f(q[i] >> 16) * 0.125f;
      q[i] = (unsigned int)f2bf(lo) | ((unsigned int)f2bf(hi) << 16);
    }
    *(vu4*)&Ql[row * 72 + cp] = q;
    vu4 kk = *(const vu4*)(K + g);
    *(vu4*)&Kl[row * 72 + cp] = kk;
    vu4 vv = *(const vu4*)(V + g);
    #pragma unroll
    for (int i = 0; i < 4; i++) {
      Vt[(cp + 2 * i)     * 136 + row] = (ushort_t)(vv[i] & 0xffffu);
      Vt[(cp + 2 * i + 1) * 136 + row] = (ushort_t)(vv[i] >> 16);
    }
  }
  __syncthreads();
  int w = t >> 5, lane = t & 31;
  int m0 = w * 16;
  int lhalf = (lane & 16) ? 8 : 0;
  int lcol  = lane & 15;
  // scores: 16 rows x 128 cols per wave
  vf8 sc[8];
  #pragma unroll
  for (int nt = 0; nt < 8; nt++) sc[nt] = (vf8){0.f,0.f,0.f,0.f,0.f,0.f,0.f,0.f};
  #pragma unroll
  for (int ks = 0; ks < 64; ks += 32) {
    vbf16 a = lds_a_frag(Ql, 72, m0, ks);
    #pragma unroll
    for (int nt = 0; nt < 8; nt++)
      sc[nt] = wmma_bf16(a, lds_b_frag(Kl, 72, nt * 16, ks), sc[nt]);
  }
  // softmax over N (spread across 8 tiles x 16 lanes of this half-wave)
  float pout[8][8];
  #pragma unroll
  for (int j = 0; j < 8; j++) {
    float m = -1e30f;
    #pragma unroll
    for (int nt = 0; nt < 8; nt++) m = fmaxf(m, sc[nt][j]);
    #pragma unroll
    for (int msk = 1; msk < 16; msk <<= 1) m = fmaxf(m, __shfl_xor(m, msk, 32));
    float s = 0.f;
    #pragma unroll
    for (int nt = 0; nt < 8; nt++) { float e = __expf(sc[nt][j] - m); pout[j][nt] = e; s += e; }
    #pragma unroll
    for (int msk = 1; msk < 16; msk <<= 1) s += __shfl_xor(s, msk, 32);
    float inv = 1.0f / s;
    #pragma unroll
    for (int nt = 0; nt < 8; nt++) pout[j][nt] *= inv;
  }
  __syncthreads();   // everyone done reading Ql/Kl
  #pragma unroll
  for (int j = 0; j < 8; j++) {
    int row = m0 + j + lhalf;
    #pragma unroll
    for (int nt = 0; nt < 8; nt++)
      Pl[row * 136 + nt * 16 + lcol] = f2bf(pout[j][nt]);
  }
  __syncthreads();
  // O = P @ V : 16 rows x 64 cols per wave
  vf8 oc[4];
  #pragma unroll
  for (int nt = 0; nt < 4; nt++) oc[nt] = (vf8){0.f,0.f,0.f,0.f,0.f,0.f,0.f,0.f};
  #pragma unroll
  for (int ks = 0; ks < 128; ks += 32) {
    vbf16 a = lds_a_frag(Pl, 136, m0, ks);
    #pragma unroll
    for (int nt = 0; nt < 4; nt++)
      oc[nt] = wmma_bf16(a, lds_b_frag(Vt, 136, nt * 16, ks), oc[nt]);
  }
  #pragma unroll
  for (int nt = 0; nt < 4; nt++)
    #pragma unroll
    for (int j = 0; j < 8; j++) {
      int row = m0 + j + lhalf;
      O[base + (size_t)row * (Hh * HDd) + nt * 16 + lcol] = f2bf(oc[nt][j]);
    }
}

// ---------------- host ----------------
extern "C" void kernel_launch(void* const* d_in, const int* in_sizes, int n_in,
                              void* d_out, int out_size, void* d_ws, size_t ws_size,
                              hipStream_t stream) {
  const float* x     = (const float*)d_in[0];
  const int*   t     = (const int*)d_in[1];
  const float* obj   = (const float*)d_in[2];
  const int*   ec    = (const int*)d_in[3];
  const int*   rel   = (const int*)d_in[4];
  const float* basis = (const float*)d_in[5];
  const float* comb  = (const float*)d_in[6];
  const float* root  = (const float*)d_in[7];
  const float* rbias = (const float*)d_in[8];
  const float* Wq_s = (const float*)d_in[9];  const float* bq_s = (const float*)d_in[10];
  const float* Wk_s = (const float*)d_in[11]; const float* bk_s = (const float*)d_in[12];
  const float* Wv_s = (const float*)d_in[13]; const float* bv_s = (const float*)d_in[14];
  const float* Wo_s = (const float*)d_in[15]; const float* bo_s = (const float*)d_in[16];
  const float* Wq_c = (const float*)d_in[17]; const float* bq_c = (const float*)d_in[18];
  const float* Wk_c = (const float*)d_in[19]; const float* bk_c = (const float*)d_in[20];
  const float* Wv_c = (const float*)d_in[21]; const float* bv_c = (const float*)d_in[22];
  const float* Wo_c = (const float*)d_in[23]; const float* bo_c = (const float*)d_in[24];
  float* out = (float*)d_out;

  char* ws = (char*)d_ws;
  size_t off = 0;
  auto alloc = [&](size_t bytes) -> char* {
    char* p = ws + off;
    off += (bytes + 255) & ~(size_t)255;
    return p;
  };
  ushort_t* S      = (ushort_t*)alloc((size_t)NN * Rr * Dd * 2);      // 128MB
  float*    hA     = (float*)alloc((size_t)NN * Dd * 4);
  float*    hB     = (float*)alloc((size_t)NN * Dd * 4);
  ushort_t* hbA    = (ushort_t*)alloc((size_t)NN * Dd * 2);
  ushort_t* hbB    = (ushort_t*)alloc((size_t)NN * Dd * 2);
  ushort_t* objbf  = (ushort_t*)alloc((size_t)NN * Cc * 2);
  ushort_t* Wcat   = (ushort_t*)alloc((size_t)Rr * Dd * Dd * 2);
  ushort_t* rootbf = (ushort_t*)alloc((size_t)Dd * Dd * 2);
  ushort_t* wqs = (ushort_t*)alloc((size_t)Dd * Dd * 2);
  ushort_t* wks = (ushort_t*)alloc((size_t)Dd * Dd * 2);
  ushort_t* wvs = (ushort_t*)alloc((size_t)Dd * Dd * 2);
  ushort_t* wos = (ushort_t*)alloc((size_t)Dd * Dd * 2);
  ushort_t* wqc = (ushort_t*)alloc((size_t)Dd * Dd * 2);
  ushort_t* wkc = (ushort_t*)alloc((size_t)Cc * Dd * 2);
  ushort_t* wvc = (ushort_t*)alloc((size_t)Cc * Dd * 2);
  ushort_t* woc = (ushort_t*)alloc((size_t)Dd * Dd * 2);
  ushort_t* Qs = (ushort_t*)alloc((size_t)NN * Dd * 2);
  ushort_t* Ks = (ushort_t*)alloc((size_t)NN * Dd * 2);
  ushort_t* Vs = (ushort_t*)alloc((size_t)NN * Dd * 2);
  ushort_t* Qc = (ushort_t*)alloc((size_t)NN * Dd * 2);
  ushort_t* Kc = (ushort_t*)alloc((size_t)NN * Dd * 2);
  ushort_t* Vc = (ushort_t*)alloc((size_t)NN * Dd * 2);
  ushort_t* Os = (ushort_t*)alloc((size_t)NN * Dd * 2);
  ushort_t* Oc = (ushort_t*)alloc((size_t)NN * Dd * 2);
  int* cnt   = (int*)alloc((size_t)SEGS * 4);
  int* offs  = (int*)alloc((size_t)(SEGS + 1) * 4);
  int* fill  = (int*)alloc((size_t)SEGS * 4);
  int* elist = (int*)alloc((size_t)Ee * 4);
  float* zbias = (float*)alloc((size_t)Dd * 4);

  auto gemm = [&](const ushort_t* A, const ushort_t* Bm, float* C, const float* bias,
                  ushort_t* Cbf, int M, int N, int K, int acc, int act) {
    dim3 g(N / 128, M / 128);
    const float* bb = bias ? bias : zbias;
    if (Cbf) {
      if (acc) { if (act) k_gemm<1,1,1><<<g,256,0,stream>>>(A,Bm,C,bb,Cbf,M,N,K);
                 else     k_gemm<1,0,1><<<g,256,0,stream>>>(A,Bm,C,bb,Cbf,M,N,K); }
      else     { if (act) k_gemm<0,1,1><<<g,256,0,stream>>>(A,Bm,C,bb,Cbf,M,N,K);
                 else     k_gemm<0,0,1><<<g,256,0,stream>>>(A,Bm,C,bb,Cbf,M,N,K); }
    } else {
      if (acc) { if (act) k_gemm<1,1,0><<<g,256,0,stream>>>(A,Bm,C,bb,Cbf,M,N,K);
                 else     k_gemm<1,0,0><<<g,256,0,stream>>>(A,Bm,C,bb,Cbf,M,N,K); }
      else     { if (act) k_gemm<0,1,0><<<g,256,0,stream>>>(A,Bm,C,bb,Cbf,M,N,K);
                 else     k_gemm<0,0,0><<<g,256,0,stream>>>(A,Bm,C,bb,Cbf,M,N,K); }
    }
  };

  // preprocessing + CSR
  k_zero_f32<<<2, 256, 0, stream>>>(zbias, Dd);
  k_pre<<<NN, 256, 0, stream>>>(x, t, obj, hA, hbA);
  k_f2bf<<<(NN * Cc + 255) / 256, 256, 0, stream>>>(obj, objbf, NN * Cc);
  k_zero_i32<<<(SEGS + 255) / 256, 256, 0, stream>>>(cnt, SEGS);
  k_zero_i32<<<(SEGS + 255) / 256, 256, 0, stream>>>(fill, SEGS);
  k_count<<<Ee / 256, 256, 0, stream>>>(ec, rel, cnt);
  k_scan<<<1, 1024, 0, stream>>>(cnt, offs, SEGS);
  k_fill<<<Ee / 256, 256, 0, stream>>>(ec, rel, offs, fill, elist);
  k_sortseg<<<SEGS / 256, 256, 0, stream>>>(offs, elist);

  // weight conversions (bf16)
  k_f2bf<<<(Dd * Dd + 255) / 256, 256, 0, stream>>>(Wq_s, wqs, Dd * Dd);
  k_f2bf<<<(Dd * Dd + 255) / 256, 256, 0, stream>>>(Wk_s, wks, Dd * Dd);
  k_f2bf<<<(Dd * Dd + 255) / 256, 256, 0, stream>>>(Wv_s, wvs, Dd * Dd);
  k_f2bf<<<(Dd * Dd + 255) / 256, 256, 0, stream>>>(Wo_s, wos, Dd * Dd);
  k_f2bf<<<(Dd * Dd + 255) / 256, 256, 0, stream>>>(Wq_c, wqc, Dd * Dd);
  k_f2bf<<<(Cc * Dd + 255) / 256, 256, 0, stream>>>(Wk_c, wkc, Cc * Dd);
  k_f2bf<<<(Cc * Dd + 255) / 256, 256, 0, stream>>>(Wv_c, wvc, Cc * Dd);
  k_f2bf<<<(Dd * Dd + 255) / 256, 256, 0, stream>>>(Wo_c, woc, Dd * Dd);

  // RGCN layers: agg-first formulation -> one big GEMM per layer + root GEMM
  float*    hcur = hA;  float*    hnext = hB;
  ushort_t* bcur = hbA; ushort_t* bnext = hbB;
  for (int l = 0; l < Ll; l++) {
    k_wcat<<<(Rr * Dd * Dd) / 256, 256, 0, stream>>>(comb + (size_t)l * Rr * NBb,
                                                     basis + (size_t)l * NBb * Dd * Dd, Wcat);
    k_f2bf<<<(Dd * Dd + 255) / 256, 256, 0, stream>>>(root + (size_t)l * Dd * Dd, rootbf, Dd * Dd);
    k_aggregate<<<SEGS / 2, 256, 0, stream>>>(hcur, offs, elist, S);
    gemm(S, Wcat, hnext, rbias + l * Dd, nullptr, NN, Dd, Rr * Dd, 0, 0);
    gemm(bcur, rootbf, hnext, nullptr, bnext, NN, Dd, Dd, 1, (l < Ll - 1) ? 1 : 0);
    float* tf = hcur; hcur = hnext; hnext = tf;
    ushort_t* tb = bcur; bcur = bnext; bnext = tb;
  }
  // bcur = rg (bf16); hnext = free fp32 scratch
  float* tmpC = hnext;

  // attention projections
  gemm(bcur,  wqs, tmpC, bq_s, Qs, NN, Dd, Dd, 0, 0);
  gemm(bcur,  wks, tmpC, bk_s, Ks, NN, Dd, Dd, 0, 0);
  gemm(bcur,  wvs, tmpC, bv_s, Vs, NN, Dd, Dd, 0, 0);
  gemm(bcur,  wqc, tmpC, bq_c, Qc, NN, Dd, Dd, 0, 0);
  gemm(objbf, wkc, tmpC, bk_c, Kc, NN, Dd, Cc, 0, 0);
  gemm(objbf, wvc, tmpC, bv_c, Vc, NN, Dd, Cc, 0, 0);

  // fused attention (self + cross share shapes: kv length = 128)
  k_attn<<<Bsz * Hh, 256, 0, stream>>>(Qs, Ks, Vs, Os);
  k_attn<<<Bsz * Hh, 256, 0, stream>>>(Qc, Kc, Vc, Oc);

  // output projections, accumulated into d_out
  gemm(Os, wos, out, bo_s, nullptr, NN, Dd, Dd, 0, 0);
  gemm(Oc, woc, out, bo_c, nullptr, NN, Dd, Dd, 1, 0);
}
